// GAT_56762287784606
// MI455X (gfx1250) — matrix operand order repaired and checked by reading, not compiled, not run on hardware
//
#include <hip/hip_runtime.h>
#include <hip/hip_bf16.h>

#define N_NODES 100000
#define N_EDGES 1600000
#define IN_FEAT 128
#define N_HEADS 8
#define OUT_FEAT 16   // H*F = 128

typedef __attribute__((ext_vector_type(16))) _Float16 v16h;
typedef __attribute__((ext_vector_type(8)))  float    v8f;

// ---- ordered-uint encoding for float atomic max (monotonic map f32 -> u32) ----
__device__ __forceinline__ unsigned enc_f(float f) {
    unsigned u = __float_as_uint(f);
    return (u & 0x80000000u) ? ~u : (u | 0x80000000u);
}
__device__ __forceinline__ float dec_f(unsigned u) {
    return (u & 0x80000000u) ? __uint_as_float(u & 0x7FFFFFFFu) : __uint_as_float(~u);
}
__device__ __forceinline__ float lrelu(float v) { return v > 0.0f ? v : 0.2f * v; }

// ---------------------------------------------------------------------------
// 1) Pack W[128][128] (f32, row-major) into per-lane WMMA B-fragments (f16).
//    Layout: Wfrag[kk][head][lane][e] ; n=lane&15, K=kk*32+(e<8 ? (lane>>4)*8+e : 16+(lane>>4)*8+e-8)
// ---------------------------------------------------------------------------
__global__ void gat_pack_wfrag(const float* __restrict__ W, _Float16* __restrict__ Wfrag) {
    int idx = blockIdx.x * blockDim.x + threadIdx.x;           // 4*8*32*16 = 16384
    if (idx >= 4 * 8 * 32 * 16) return;
    int e    = idx & 15;
    int lane = (idx >> 4) & 31;
    int h    = (idx >> 9) & 7;
    int kk   = idx >> 12;
    int n    = lane & 15;
    int k    = kk * 32 + ((e < 8) ? ((lane >> 4) * 8 + e) : (16 + (lane >> 4) * 8 + (e - 8)));
    Wfrag[idx] = (_Float16)W[k * IN_FEAT + h * OUT_FEAT + n];
}

// ---------------------------------------------------------------------------
// 2) proj = X @ W via v_wmma_f32_16x16x32_f16.
//    Block = 256 threads = 8 waves; block -> 16-row M tile; wave w -> head w.
// ---------------------------------------------------------------------------
__global__ void __launch_bounds__(256)
gat_proj_wmma(const float* __restrict__ X, const _Float16* __restrict__ Wfrag,
              float* __restrict__ proj) {
    const int wave = threadIdx.x >> 5;       // head 0..7
    const int lane = threadIdx.x & 31;
    const int m0   = blockIdx.x * 16;
    const int row  = m0 + (lane & 15);
    const int kb   = (lane >> 4) * 8;        // A-fragment K sub-offset per half-wave

    v8f c = {};
#pragma unroll
    for (int kk = 0; kk < 4; ++kk) {
        const float* xr = X + row * IN_FEAT + kk * 32 + kb;
        v16h a;
#pragma unroll
        for (int i = 0; i < 8; ++i) a[i]     = (_Float16)xr[i];
#pragma unroll
        for (int i = 0; i < 8; ++i) a[8 + i] = (_Float16)xr[16 + i];
        const v16h b = *(const v16h*)(Wfrag + (((kk * 8 + wave) * 32) + lane) * 16);
        c = __builtin_amdgcn_wmma_f32_16x16x32_f16(false, a, false, b,
                                                   (short)0, c, false, false);
    }
    // C/D layout: VGPR r, lanes 0-15 -> M=r, lanes 16-31 -> M=r+8; N = lane&15
    const int n     = wave * OUT_FEAT + (lane & 15);
    const int rbase = m0 + ((lane >> 4) * 8);
#pragma unroll
    for (int r = 0; r < 8; ++r)
        proj[(rbase + r) * IN_FEAT + n] = c[r];
}

// ---------------------------------------------------------------------------
// 3) s_src[n,h] = proj[n,h,:]·a_src[h,:]  (same for s_tgt)
// ---------------------------------------------------------------------------
__global__ void gat_scores(const float* __restrict__ proj,
                           const float* __restrict__ a_src, const float* __restrict__ a_tgt,
                           float* __restrict__ s_src, float* __restrict__ s_tgt) {
    int t = blockIdx.x * blockDim.x + threadIdx.x;
    if (t >= N_NODES * N_HEADS) return;
    int n = t >> 3, h = t & 7;
    const float* p  = proj  + n * IN_FEAT + h * OUT_FEAT;
    const float* as = a_src + h * OUT_FEAT;
    const float* at = a_tgt + h * OUT_FEAT;
    float s1 = 0.0f, s2 = 0.0f;
#pragma unroll
    for (int f = 0; f < OUT_FEAT; ++f) { float v = p[f]; s1 += v * as[f]; s2 += v * at[f]; }
    s_src[t] = s1; s_tgt[t] = s2;
}

// ---------------------------------------------------------------------------
// 4) zero accum (N*128 f32), smax (N*8 u32 = ordered -inf), denom (N*8 f32)
// ---------------------------------------------------------------------------
__global__ void gat_init(float* __restrict__ accum, unsigned* __restrict__ smax,
                         float* __restrict__ denom) {
    int t = blockIdx.x * blockDim.x + threadIdx.x;
    if (t < N_NODES * IN_FEAT) accum[t] = 0.0f;
    else if (t < N_NODES * IN_FEAT + N_NODES * N_HEADS) smax[t - N_NODES * IN_FEAT] = 0u;
    else if (t < N_NODES * IN_FEAT + 2 * N_NODES * N_HEADS)
        denom[t - N_NODES * IN_FEAT - N_NODES * N_HEADS] = 0.0f;
}

// ---------------------------------------------------------------------------
// 5) per-edge segment max of leakyrelu scores (ordered-uint atomic max)
// ---------------------------------------------------------------------------
__global__ void gat_edge_max(const int* __restrict__ src, const int* __restrict__ tgt,
                             const float* __restrict__ s_src, const float* __restrict__ s_tgt,
                             unsigned* __restrict__ smax) {
    int e = blockIdx.x * blockDim.x + threadIdx.x;
    if (e >= N_EDGES) return;
    int s = src[e], t = tgt[e];
#pragma unroll
    for (int h = 0; h < N_HEADS; ++h) {
        float v = lrelu(s_src[s * N_HEADS + h] + s_tgt[t * N_HEADS + h]);
        atomicMax(&smax[t * N_HEADS + h], enc_f(v));
    }
}

// ---------------------------------------------------------------------------
// 6) per-edge exp-sum into denom
// ---------------------------------------------------------------------------
__global__ void gat_edge_sum(const int* __restrict__ src, const int* __restrict__ tgt,
                             const float* __restrict__ s_src, const float* __restrict__ s_tgt,
                             const unsigned* __restrict__ smax, float* __restrict__ denom) {
    int e = blockIdx.x * blockDim.x + threadIdx.x;
    if (e >= N_EDGES) return;
    int s = src[e], t = tgt[e];
#pragma unroll
    for (int h = 0; h < N_HEADS; ++h) {
        float v  = lrelu(s_src[s * N_HEADS + h] + s_tgt[t * N_HEADS + h]);
        float m  = dec_f(smax[t * N_HEADS + h]);
        atomicAdd(&denom[t * N_HEADS + h], __expf(v - m));
    }
}

// ---------------------------------------------------------------------------
// 7) per-(edge,head): alpha * proj[src] scatter-added into accum[tgt]
// ---------------------------------------------------------------------------
__global__ void gat_edge_msg(const int* __restrict__ src, const int* __restrict__ tgt,
                             const float* __restrict__ s_src, const float* __restrict__ s_tgt,
                             const unsigned* __restrict__ smax, const float* __restrict__ denom,
                             const float* __restrict__ proj, float* __restrict__ accum) {
    int id = blockIdx.x * blockDim.x + threadIdx.x;
    if (id >= N_EDGES * N_HEADS) return;
    int e = id >> 3, h = id & 7;
    int s = src[e], t = tgt[e];
    float v     = lrelu(s_src[s * N_HEADS + h] + s_tgt[t * N_HEADS + h]);
    float m     = dec_f(smax[t * N_HEADS + h]);
    float alpha = __expf(v - m) / (denom[t * N_HEADS + h] + 1e-16f);
    const float* p = proj  + s * IN_FEAT + h * OUT_FEAT;
    float*       a = accum + t * IN_FEAT + h * OUT_FEAT;
#pragma unroll
    for (int q = 0; q < 4; ++q) {
        float4 pv = ((const float4*)p)[q];
        atomicAdd(&a[4 * q + 0], pv.x * alpha);
        atomicAdd(&a[4 * q + 1], pv.y * alpha);
        atomicAdd(&a[4 * q + 2], pv.z * alpha);
        atomicAdd(&a[4 * q + 3], pv.w * alpha);
    }
}

// ---------------------------------------------------------------------------
// 8) mean over heads + bias + softmax(16)
// ---------------------------------------------------------------------------
__global__ void gat_finalize(const float* __restrict__ accum, const float* __restrict__ bias,
                             float* __restrict__ out) {
    int n = blockIdx.x * blockDim.x + threadIdx.x;
    if (n >= N_NODES) return;
    float o[OUT_FEAT];
#pragma unroll
    for (int f = 0; f < OUT_FEAT; ++f) o[f] = bias[f];
    const float* a = accum + n * IN_FEAT;
#pragma unroll
    for (int h = 0; h < N_HEADS; ++h)
#pragma unroll
        for (int f = 0; f < OUT_FEAT; ++f) o[f] += 0.125f * a[h * OUT_FEAT + f];
    float mx = o[0];
#pragma unroll
    for (int f = 1; f < OUT_FEAT; ++f) mx = fmaxf(mx, o[f]);
    float sum = 0.0f;
#pragma unroll
    for (int f = 0; f < OUT_FEAT; ++f) { o[f] = __expf(o[f] - mx); sum += o[f]; }
    float inv = 1.0f / sum;
#pragma unroll
    for (int f = 0; f < OUT_FEAT; ++f) out[n * OUT_FEAT + f] = o[f] * inv;
}

extern "C" void kernel_launch(void* const* d_in, const int* in_sizes, int n_in,
                              void* d_out, int out_size, void* d_ws, size_t ws_size,
                              hipStream_t stream) {
    const float* X     = (const float*)d_in[0];
    const int*   eidx  = (const int*)d_in[1];     // [2, E] flattened
    const float* W     = (const float*)d_in[2];
    const float* a_src = (const float*)d_in[3];
    const float* a_tgt = (const float*)d_in[4];
    const float* bias  = (const float*)d_in[5];
    float*       out   = (float*)d_out;

    const int* src = eidx;
    const int* tgt = eidx + N_EDGES;

    // workspace layout (f32 element offsets)
    float* ws = (float*)d_ws;
    float*    proj  = ws;                                   // 12,800,000
    float*    accum = ws + 12800000;                        // 12,800,000
    float*    s_src = ws + 25600000;                        //    800,000
    float*    s_tgt = ws + 26400000;                        //    800,000
    unsigned* smax  = (unsigned*)(ws + 27200000);           //    800,000
    float*    denom = ws + 28000000;                        //    800,000
    _Float16* wfrag = (_Float16*)(ws + 28800000);           //     16,384 halfs

    const int T = 256;
    gat_pack_wfrag<<<(4 * 8 * 32 * 16 + T - 1) / T, T, 0, stream>>>(W, wfrag);
    gat_proj_wmma<<<N_NODES / 16, T, 0, stream>>>(X, wfrag, proj);
    gat_scores<<<(N_NODES * N_HEADS + T - 1) / T, T, 0, stream>>>(proj, a_src, a_tgt, s_src, s_tgt);
    {
        int cnt = N_NODES * IN_FEAT + 2 * N_NODES * N_HEADS;
        gat_init<<<(cnt + T - 1) / T, T, 0, stream>>>(accum, smax, denom);
    }
    gat_edge_max<<<(N_EDGES + T - 1) / T, T, 0, stream>>>(src, tgt, s_src, s_tgt, smax);
    gat_edge_sum<<<(N_EDGES + T - 1) / T, T, 0, stream>>>(src, tgt, s_src, s_tgt, smax, denom);
    gat_edge_msg<<<(N_EDGES * N_HEADS + T - 1) / T, T, 0, stream>>>(src, tgt, s_src, s_tgt,
                                                                    smax, denom, proj, accum);
    gat_finalize<<<(N_NODES + T - 1) / T, T, 0, stream>>>(accum, bias, out);
}
